// MHSelfAttention_20753281974757
// MI455X (gfx1250) — compile-verified, matrix operand order checked by reading, not run on hardware
//
#include <hip/hip_runtime.h>
#include <hip/hip_bf16.h>

#define B_    4
#define L_    1536
#define D_    1536
#define H_    8
#define DQK_  64
#define DV_   192
#define QKP_  512      // H_*DQK_
#define VP_   1536     // H_*DV_
#define NQKV_ 2560     // QKP_+QKP_+VP_
#define PF_   192      // POS_FEATS
#define TL_   3071     // 2L-1
#define TLP_  3072     // padded
#define ROWS_ (B_*L_)  // 6144

#if __has_builtin(__builtin_amdgcn_global_load_async_to_lds_b128) && \
    __has_builtin(__builtin_amdgcn_s_wait_asynccnt)
#define USE_ASYNC_LDS 1
#else
#define USE_ASYNC_LDS 0
#endif

typedef _Float16 v16h __attribute__((ext_vector_type(16)));
typedef _Float16 v8h  __attribute__((ext_vector_type(8)));
typedef float    v8f  __attribute__((ext_vector_type(8)));
typedef int      v4i  __attribute__((ext_vector_type(4)));
typedef __attribute__((address_space(1))) v4i* gas_v4i_ptr;  // global
typedef __attribute__((address_space(3))) v4i* las_v4i_ptr;  // LDS

// ---- 16x32 f16 A/B fragment loader (ISA 7.12.2 16-bit layout) ------------
// lane group g = lane>>4 selects K-halves: elems 0..7 <- K = 8g..8g+7,
// elems 8..15 <- K = 16+8g..16+8g+7. p points at the lane's 32-elem row chunk.
__device__ inline v16h load_frag32(const _Float16* p, int g) {
  v8h lo = *(const v8h*)(p + 8 * g);
  v8h hi = *(const v8h*)(p + 16 + 8 * g);
  v16h r;
#pragma unroll
  for (int i = 0; i < 8; ++i) { r[i] = lo[i]; r[i + 8] = hi[i]; }
  return r;
}

__device__ inline v8f wmma32(v16h a, v16h b, v8f c) {
  return __builtin_amdgcn_wmma_f32_16x16x32_f16(false, a, false, b, (short)0, c,
                                                false, false);
}

// wave32: row of a C-fragment lives in 16 contiguous lanes -> xor masks 1..8
__device__ inline float rmax16(float v) {
#pragma unroll
  for (int m = 1; m < 16; m <<= 1) v = fmaxf(v, __shfl_xor(v, m, 32));
  return v;
}
__device__ inline float rsum16(float v) {
#pragma unroll
  for (int m = 1; m < 16; m <<= 1) v += __shfl_xor(v, m, 32);
  return v;
}

// ---------------------------- prep kernels --------------------------------
__global__ void k_f32_to_f16(const float* __restrict__ s, _Float16* __restrict__ d, int n) {
  int i = blockIdx.x * 256 + threadIdx.x;
  if (i < n) d[i] = (_Float16)s[i];
}

// dst[n*K+k] = src[k*N+n]  (f32 -> f16 transpose)
__global__ void k_transpose_f16(const float* __restrict__ src, _Float16* __restrict__ dst,
                                int K, int N) {
  int i = blockIdx.x * 256 + threadIdx.x;
  if (i < K * N) {
    int k = i / N, n = i % N;
    dst[(size_t)n * K + k] = (_Float16)src[i];
  }
}

// fused transposed weights: rows 0..511 = WQ^T, 512..1023 = WK^T, 1024..2559 = WV^T
__global__ void k_build_wqkvT(const float* __restrict__ WQ, const float* __restrict__ WK,
                              const float* __restrict__ WV, _Float16* __restrict__ dst) {
  int i = blockIdx.x * 256 + threadIdx.x;
  if (i < NQKV_ * D_) {
    int n = i / D_, k = i % D_;
    float v = (n < QKP_) ? WQ[(size_t)k * QKP_ + n]
            : (n < 2 * QKP_) ? WK[(size_t)k * QKP_ + (n - QKP_)]
                             : WV[(size_t)k * VP_ + (n - 2 * QKP_)];
    dst[i] = (_Float16)v;
  }
}

// gamma basis (unnormalized) + global max (order-independent atomicMax on bits)
__global__ void k_gamma(float* __restrict__ gp, unsigned* __restrict__ gmax) {
  int i = blockIdx.x * 256 + threadIdx.x;
  float v = 0.f;
  if (i < TL_ * 32) {
    int pos = i >> 5, f = i & 31;
    float pa = fabsf((float)(pos - (L_ - 1)));
    const float stddev = (float)L_ / 64.f;           // 24
    float mean = ((float)L_ / 32.f) + ((float)L_ - (float)L_ / 32.f) * (float)f / 31.f;
    float conc = (mean / stddev) * (mean / stddev);
    float rate = mean / (stddev * stddev);
    float lu = (pa > 0.f) ? (conc - 1.f) * logf(pa) - rate * pa : -1e30f; // xlogy, conc>1
    float ln = lgammaf(conc) - conc * logf(rate);
    v = __expf(lu - ln) + 1e-8f;
    gp[i] = v;
  }
  __shared__ float red[256];
  red[threadIdx.x] = v;
  __syncthreads();
  for (int s = 128; s > 0; s >>= 1) {
    if (threadIdx.x < s) red[threadIdx.x] = fmaxf(red[threadIdx.x], red[threadIdx.x + s]);
    __syncthreads();
  }
  if (threadIdx.x == 0) atomicMax(gmax, __float_as_uint(red[0]));
}

// full positional features [TLP_,192] f16; padded row 3071 is zero
__global__ void k_pos(const float* __restrict__ gp, const float* __restrict__ gmax,
                      _Float16* __restrict__ posh) {
  int i = blockIdx.x * 256 + threadIdx.x;
  if (i >= TLP_ * 96) return;
  int pos = i / 96, k = i % 96;
  if (pos >= TL_) {
    posh[(size_t)pos * PF_ + k] = (_Float16)0.f;
    posh[(size_t)pos * PF_ + 96 + k] = (_Float16)0.f;
    return;
  }
  float p = (float)(pos - (L_ - 1));
  float pa = fabsf(p);
  int f = k & 31;
  float e;
  if (k < 32) {                                       // exponential
    float max_range = logf((float)L_) / logf(2.f);
    float hl = exp2f(3.f + (max_range - 3.f) * (float)f / 31.f);
    e = __expf(-0.6931471805599453f / hl * pa);
  } else if (k < 64) {                                // central mask
    float cw = exp2f((float)(f + 1)) - 1.f;
    e = (cw > pa) ? 1.f : 0.f;
  } else {                                            // gamma (normalized)
    e = gp[pos * 32 + f] / gmax[0];
  }
  float sgn = (p > 0.f) ? 1.f : (p < 0.f ? -1.f : 0.f);
  posh[(size_t)pos * PF_ + k] = (_Float16)e;
  posh[(size_t)pos * PF_ + 96 + k] = (_Float16)(sgn * e);
}

// Qw = Q*dqk^-0.5 + r_w_bias ; Qr = Q*dqk^-0.5 + r_r_bias   (f16, [ROWS_,512])
__global__ void k_biasq(const _Float16* __restrict__ qkv, const float* __restrict__ rwb,
                        const float* __restrict__ rrb, _Float16* __restrict__ qw,
                        _Float16* __restrict__ qr) {
  int i = blockIdx.x * 256 + threadIdx.x;
  if (i < ROWS_ * QKP_) {
    int row = i >> 9, c = i & (QKP_ - 1);
    float q = (float)qkv[(size_t)row * NQKV_ + c] * 0.125f; // 64^-0.5
    qw[i] = (_Float16)(q + rwb[c]);
    qr[i] = (_Float16)(q + rrb[c]);
  }
}

// V transpose to [B,H,192,L] f16 via LDS tiles (coalesced both sides)
__global__ void k_vt(const _Float16* __restrict__ qkv, _Float16* __restrict__ vt) {
  __shared__ _Float16 t[32][33];
  int tx = threadIdx.x & 31, ty = threadIdx.x >> 5;   // 32 x 8
  int jt = blockIdx.x * 32;
  int nt = (blockIdx.y % (DV_ / 32)) * 32;
  int bh = blockIdx.y / (DV_ / 32);
  int b = bh >> 3, h = bh & 7;
#pragma unroll
  for (int s = 0; s < 32; s += 8) {
    int j = jt + ty + s;
    t[ty + s][tx] = qkv[(size_t)(b * L_ + j) * NQKV_ + 2 * QKP_ + h * DV_ + nt + tx];
  }
  __syncthreads();
#pragma unroll
  for (int s = 0; s < 32; s += 8) {
    int n = nt + ty + s;
    vt[((size_t)(b * H_ + h) * DV_ + n) * L_ + jt + tx] = t[tx][ty + s];
  }
}

// ----------------------- generic WMMA GEMM --------------------------------
// C[M,N] = A[M,K] * BT[N,K]^T ; block = 4 waves (64 rows) x 128 cols.
// The 128x32 B-tile is staged to LDS once per k-step (async-to-LDS with
// double buffering when the gfx1250 builtins exist) and shared by all waves:
// 4x less L2 traffic on B, loads overlap the 32 WMMAs of the previous tile.
// OUT_MODE 0: f16 out, 1: f32 out + bias.
template <int OUT_MODE>
__global__ __launch_bounds__(128, 1) void k_gemm(
    const _Float16* __restrict__ A, const _Float16* __restrict__ BT,
    _Float16* __restrict__ Ch, float* __restrict__ Cf,
    const float* __restrict__ bias, int M, int N, int K) {
  const int t = threadIdx.x;
  const int lane = t & 31;
  const int wave = t >> 5;
  const int g = lane >> 4, nl = lane & 15;
  const int m0 = blockIdx.y * 64 + wave * 16;
  const int n0 = blockIdx.x * 128;

  __shared__ __align__(16) _Float16 bt[2][128 * 32];  // [buf][n][k] 8KB each

  // stage the 128x32 B tile for k-chunk k0 into bt[buf]: thread t owns row n0+t
  auto stage = [&](int buf, int k0) {
    const _Float16* gsrc = BT + (size_t)(n0 + t) * K + k0;
    _Float16* ldst = &bt[buf][t * 32];
#if USE_ASYNC_LDS
#pragma unroll
    for (int c = 0; c < 4; ++c)
      __builtin_amdgcn_global_load_async_to_lds_b128(
          (gas_v4i_ptr)(gsrc + 8 * c), (las_v4i_ptr)(ldst + 8 * c), 0, 0);
#else
#pragma unroll
    for (int c = 0; c < 4; ++c)
      ((v8h*)ldst)[c] = ((const v8h*)gsrc)[c];
#endif
  };

  v8f acc[8] = {};
  const _Float16* arow = A + (size_t)(m0 + nl) * K;

  stage(0, 0);
#if USE_ASYNC_LDS
  __builtin_amdgcn_s_wait_asynccnt(0);
#endif
  __syncthreads();

  int cur = 0;
  for (int k0 = 0; k0 < K; k0 += 32) {
    if (k0 + 32 < K) stage(cur ^ 1, k0 + 32);
    if (k0 + 256 < K) __builtin_prefetch(arow + k0 + 256, 0, 0);
    v16h a = load_frag32(arow + k0, g);
#pragma unroll
    for (int f = 0; f < 8; ++f) {
      v16h bfr = load_frag32(&bt[cur][(f * 16 + nl) * 32], g);
      acc[f] = wmma32(a, bfr, acc[f]);
    }
#if USE_ASYNC_LDS
    __builtin_amdgcn_s_wait_asynccnt(0);
#endif
    __syncthreads();
    cur ^= 1;
  }
#pragma unroll
  for (int f = 0; f < 8; ++f) {
#pragma unroll
    for (int r = 0; r < 8; ++r) {
      size_t m = (size_t)m0 + r + 8 * g;
      size_t n = (size_t)n0 + f * 16 + nl;
      if (OUT_MODE == 0) Ch[m * N + n] = (_Float16)acc[f][r];
      else               Cf[m * N + n] = acc[f][r] + bias[n];
    }
  }
}

// -------------------- fused relative flash attention ----------------------
// 1 wave = one 16-query tile of one (b,h). Block = 4 waves.
// Rel-logit tiles are Toeplitz: consecutive 16-key subtiles reuse the second
// 16-column rel tile as the next subtile's first -> 2 (not 4) WMMAs/subtile.
// Per 32-key chunk: ~4 WMMA rel + 4 WMMA content (rel as C operand via LDS
// diagonal gather) + online softmax + 12 WMMA P*V into O[16x192].
__global__ __launch_bounds__(128, 1) void k_attn(
    const _Float16* __restrict__ qw, const _Float16* __restrict__ qr,
    const _Float16* __restrict__ qkv, const _Float16* __restrict__ rk,
    const _Float16* __restrict__ vt, _Float16* __restrict__ att) {
  const int lane = threadIdx.x & 31;
  const int wave = threadIdx.x >> 5;
  const int g = lane >> 4, nl = lane & 15;
  const int tile = blockIdx.x * 4 + wave;
  const int qtiles = L_ / 16;
  const int bh = tile / qtiles;
  const int i0 = (tile % qtiles) * 16;
  const int b = bh / H_, h = bh % H_;

  __shared__ float sscr[4][16 * 32];                    // rel-shift scratch (f32)
  __shared__ __align__(16) _Float16 sp[4][16 * 32];     // P repack (f16)
  float* scr = sscr[wave];
  _Float16* ph = sp[wave];

  const _Float16* qwrow = qw + (size_t)(b * L_ + i0 + nl) * QKP_ + h * DQK_;
  const _Float16* qrrow = qr + (size_t)(b * L_ + i0 + nl) * QKP_ + h * DQK_;
  v16h aw0 = load_frag32(qwrow, g), aw1 = load_frag32(qwrow + 32, g);
  v16h ar0 = load_frag32(qrrow, g), ar1 = load_frag32(qrrow + 32, g);

  // 16x16 rel-logit tile: (Q+r_r_bias) x rK[tbase..tbase+15]^T
  auto tile_rel = [&](int tbase) -> v8f {
    const _Float16* r0 = rk + (size_t)(tbase + nl) * QKP_ + h * DQK_;
    v8f acc = {};
    v16h b0 = load_frag32(r0, g), b1 = load_frag32(r0 + 32, g);
    acc = wmma32(ar0, b0, acc);
    acc = wmma32(ar1, b1, acc);
    return acc;
  };

  v8f o[12] = {};
  float mrow[8], lrow[8];
#pragma unroll
  for (int r = 0; r < 8; ++r) { mrow[r] = -1e30f; lrow[r] = 0.f; }

  // rel window base for js=0 is (L-1)-i0-15 >= 0; last tile top row is
  // (L-1)-i0+1+(L-16) <= 2L-16, so rows stay inside the 3072-row padded rK.
  v8f relA = tile_rel((L_ - 1) - i0 - 15);

  for (int j0 = 0; j0 < L_; j0 += 32) {
    v8f sfr[2];
#pragma unroll
    for (int s = 0; s < 2; ++s) {
      const int js = j0 + 16 * s;
      v8f relB = tile_rel((L_ - 1) - i0 + 1 + js);      // next 16-col window
#pragma unroll
      for (int r = 0; r < 8; ++r) {
        scr[(r + 8 * g) * 32 + nl] = relA[r];
        scr[(r + 8 * g) * 32 + 16 + nl] = relB[r];
      }
      // relative_shift gather: logits[m][n] uses rel[m][15+n-m] (DS in-order per wave)
      v8f c;
#pragma unroll
      for (int r = 0; r < 8; ++r) {
        int m = r + 8 * g;
        c[r] = scr[m * 32 + 15 + nl - m];
      }
      // content logits on top of rel C-operand
      const _Float16* kr = qkv + (size_t)(b * L_ + js + nl) * NQKV_ + QKP_ + h * DQK_;
      v16h kb0 = load_frag32(kr, g), kb1 = load_frag32(kr + 32, g);
      c = wmma32(aw0, kb0, c);
      c = wmma32(aw1, kb1, c);
      sfr[s] = c;
      relA = relB;                                      // Toeplitz reuse
    }
    // ---- online softmax over the 32-key chunk ----
    float nm[8], sc[8];
#pragma unroll
    for (int r = 0; r < 8; ++r) {
      float mx = rmax16(fmaxf(sfr[0][r], sfr[1][r]));
      nm[r] = fmaxf(mrow[r], mx);
      sc[r] = __expf(mrow[r] - nm[r]);
      mrow[r] = nm[r];
    }
#pragma unroll
    for (int r = 0; r < 8; ++r) {
      float p0 = __expf(sfr[0][r] - nm[r]);
      float p1 = __expf(sfr[1][r] - nm[r]);
      ph[(r + 8 * g) * 32 + nl] = (_Float16)p0;
      ph[(r + 8 * g) * 32 + 16 + nl] = (_Float16)p1;
      lrow[r] = lrow[r] * sc[r] + rsum16(p0 + p1);
    }
#pragma unroll
    for (int f = 0; f < 12; ++f)
#pragma unroll
      for (int r = 0; r < 8; ++r) o[f][r] *= sc[r];
    // ---- P (A-frag via LDS repack) x V^T rows: 12 WMMA -> O[16x192] ----
    v16h pfr = load_frag32(ph + nl * 32, g);
#pragma unroll
    for (int f = 0; f < 12; ++f) {
      const _Float16* vr = vt + ((size_t)(b * H_ + h) * DV_ + f * 16 + nl) * L_ + j0;
      v16h vb = load_frag32(vr, g);
      o[f] = wmma32(pfr, vb, o[f]);
    }
  }
  // ---- finalize: attended[b, i, h*192+n] = O/l ----
#pragma unroll
  for (int f = 0; f < 12; ++f) {
#pragma unroll
    for (int r = 0; r < 8; ++r) {
      size_t row = (size_t)b * L_ + i0 + r + 8 * g;
      att[row * VP_ + h * DV_ + f * 16 + nl] = (_Float16)(o[f][r] / lrow[r]);
    }
  }
}

// ------------------------------- launch -----------------------------------
extern "C" void kernel_launch(void* const* d_in, const int* in_sizes, int n_in,
                              void* d_out, int out_size, void* d_ws, size_t ws_size,
                              hipStream_t stream) {
  const float* x    = (const float*)d_in[0];
  const float* WQ   = (const float*)d_in[1];
  const float* WK   = (const float*)d_in[2];
  const float* WV   = (const float*)d_in[3];
  const float* Wrel = (const float*)d_in[4];
  const float* Wout = (const float*)d_in[5];
  const float* bout = (const float*)d_in[6];
  const float* rwb  = (const float*)d_in[7];
  const float* rrb  = (const float*)d_in[8];
  float* out = (float*)d_out;

  char* p = (char*)d_ws;
  auto carve = [&](size_t bytes) { char* r = p; p += (bytes + 255) & ~(size_t)255; return r; };
  _Float16* xh    = (_Float16*)carve((size_t)ROWS_ * D_ * 2);
  _Float16* wqkvT = (_Float16*)carve((size_t)NQKV_ * D_ * 2);
  _Float16* wrelT = (_Float16*)carve((size_t)QKP_ * PF_ * 2);
  _Float16* woutT = (_Float16*)carve((size_t)VP_ * VP_ * 2);
  _Float16* posh  = (_Float16*)carve((size_t)TLP_ * PF_ * 2);
  _Float16* qkvh  = (_Float16*)carve((size_t)ROWS_ * NQKV_ * 2);
  _Float16* qwb   = (_Float16*)carve((size_t)ROWS_ * QKP_ * 2);
  _Float16* qrb   = (_Float16*)carve((size_t)ROWS_ * QKP_ * 2);
  _Float16* rkh   = (_Float16*)carve((size_t)TLP_ * QKP_ * 2);
  _Float16* vth   = (_Float16*)carve((size_t)B_ * H_ * DV_ * L_ * 2);
  _Float16* atth  = (_Float16*)carve((size_t)ROWS_ * VP_ * 2);
  float*    gp    = (float*)carve((size_t)TL_ * 32 * 4);
  unsigned* gmax  = (unsigned*)carve(256);

  (void)hipMemsetAsync(gmax, 0, 4, stream);
  { int n = ROWS_ * D_;  k_f32_to_f16<<<(n + 255) / 256, 256, 0, stream>>>(x, xh, n); }
  { int n = NQKV_ * D_;  k_build_wqkvT<<<(n + 255) / 256, 256, 0, stream>>>(WQ, WK, WV, wqkvT); }
  { int n = PF_ * QKP_;  k_transpose_f16<<<(n + 255) / 256, 256, 0, stream>>>(Wrel, wrelT, PF_, QKP_); }
  { int n = VP_ * VP_;   k_transpose_f16<<<(n + 255) / 256, 256, 0, stream>>>(Wout, woutT, VP_, VP_); }
  { int n = TL_ * 32;    k_gamma<<<(n + 255) / 256, 256, 0, stream>>>(gp, gmax); }
  { int n = TLP_ * 96;   k_pos<<<(n + 255) / 256, 256, 0, stream>>>(gp, (const float*)gmax, posh); }

  // QKV projection: [6144,1536] x [1536,2560]
  k_gemm<0><<<dim3(NQKV_ / 128, ROWS_ / 64), 128, 0, stream>>>(
      xh, wqkvT, qkvh, nullptr, nullptr, ROWS_, NQKV_, D_);
  // rK: [3072,192] x [192,512]
  k_gemm<0><<<dim3(QKP_ / 128, TLP_ / 64), 128, 0, stream>>>(
      posh, wrelT, rkh, nullptr, nullptr, TLP_, QKP_, PF_);

  { int n = ROWS_ * QKP_; k_biasq<<<(n + 255) / 256, 256, 0, stream>>>(qkvh, rwb, rrb, qwb, qrb); }
  k_vt<<<dim3(L_ / 32, (DV_ / 32) * B_ * H_), 256, 0, stream>>>(qkvh, vth);

  // fused relative flash attention: 3072 query tiles, 4 waves/block
  k_attn<<<(B_ * H_ * (L_ / 16)) / 4, 128, 0, stream>>>(qwb, qrb, qkvh, rkh, vth, atth);

  // output projection + bias: [6144,1536] x [1536,1536] -> f32 d_out
  k_gemm<1><<<dim3(VP_ / 128, ROWS_ / 64), 128, 0, stream>>>(
      atth, woutT, nullptr, out, bout, ROWS_, VP_, VP_);
}
